// RecurrentFormulationNetwork_1760936591509
// MI455X (gfx1250) — matrix operand order repaired and checked by reading, not compiled, not run on hardware
//
#include <hip/hip_runtime.h>
#include <hip/hip_bf16.h>
#include <math.h>
#include <stdint.h>

// ---------------------------------------------------------------------------
// RecurrentFormulationNetwork forward for MI455X (gfx1250, wave32, WMMA).
// Dense matmuls (A@A augment, A^T@Z aggregation, X@W, linears) run on a
// f16-in/f32-accumulate WMMA GEMM (v_wmma_f32_16x16x32_f16). Operands are
// pre-converted to f16 (B transposed to K-contiguous) so tiles stream into
// LDS via GLOBAL_LOAD_ASYNC_TO_LDS_B128 (ASYNCcnt) and fragments load as
// contiguous ds_load_b128s.
// ---------------------------------------------------------------------------

typedef __attribute__((ext_vector_type(16))) _Float16 v16h;
typedef __attribute__((ext_vector_type(8)))  _Float16 v8h;
typedef __attribute__((ext_vector_type(8)))  float    v8f;

#define HDIM   256
#define NNODE  4096
#define NFLD   3
#define NMESH  8
#define NBND   1
#define INDIM  12          // NFLD + NBND + NMESH
#define NTIME  3
#define DEPTH  3
#define TSTEP  (4.8f / 200.0f)

#define TM 64
#define TN 64
#define TKK 32
#define LPAD 8             // halves of row padding (keeps 16B alignment)

// ------------------------------ device helpers -----------------------------

__device__ __forceinline__ float mishf(float x) {
  float sp = (x > 20.0f) ? x : log1pf(expf(x));
  return x * tanhf(sp);
}

// ------------------------------ WMMA GEMM ----------------------------------
// C[M,N] = A @ B with A given as f16 row-major MxK and B given as f16
// row-major NxK (i.e. B transposed, K contiguous). f32 accumulate via WMMA.
__global__ __launch_bounds__(256) void k_gemm(const _Float16* __restrict__ A,
                                              const _Float16* __restrict__ BT,
                                              float* __restrict__ C,
                                              int M, int N, int K) {
  __shared__ __align__(16) _Float16 As[TM][TKK + LPAD];
  __shared__ __align__(16) _Float16 Bs[TN][TKK + LPAD];

  const int tid  = threadIdx.x;
  const int lane = tid & 31;
  const int wave = tid >> 5;                 // 0..7
  const int bm   = blockIdx.y * TM;
  const int bn   = blockIdx.x * TN;
  const int wm   = (wave >> 2) * 16;         // 0 or 16
  const int wn   = (wave & 3) * 16;          // 0,16,32,48
  const int r16  = lane & 15;
  const int grp  = lane >> 4;                // 0 or 1
  const int ci   = tid >> 2;                 // staging row 0..63
  const int cc   = (tid & 3) * 8;            // staging half-offset 0/8/16/24

  v8f acc0 = {0.f, 0.f, 0.f, 0.f, 0.f, 0.f, 0.f, 0.f};
  v8f acc1 = {0.f, 0.f, 0.f, 0.f, 0.f, 0.f, 0.f, 0.f};

  for (int k0 = 0; k0 < K; k0 += TKK) {
    const bool fullK = (k0 + TKK) <= K;

    // ---- stage A tile (TM x TKK): one async b128 per lane on fast path ----
    if (fullK && (bm + TM) <= M) {
      const _Float16* gp = A + (size_t)(bm + ci) * K + k0 + cc;
      unsigned lp = (unsigned)(size_t)&As[ci][cc];
      asm volatile("global_load_async_to_lds_b128 %0, %1, off"
                   :: "v"(lp), "v"(gp) : "memory");
    } else {
      for (int t = tid; t < TM * TKK; t += 256) {
        int i = t >> 5, kk = t & 31;
        int gm = bm + i, gk = k0 + kk;
        As[i][kk] = (gm < M && gk < K) ? A[(size_t)gm * K + gk] : (_Float16)0.0f;
      }
    }
    // ---- stage B tile (TN cols x TKK) from K-contiguous BT ----
    if (fullK && (bn + TN) <= N) {
      const _Float16* gp = BT + (size_t)(bn + ci) * K + k0 + cc;
      unsigned lp = (unsigned)(size_t)&Bs[ci][cc];
      asm volatile("global_load_async_to_lds_b128 %0, %1, off"
                   :: "v"(lp), "v"(gp) : "memory");
    } else {
      for (int t = tid; t < TN * TKK; t += 256) {
        int j = t >> 5, kk = t & 31;
        int gn = bn + j, gk = k0 + kk;
        Bs[j][kk] = (gn < N && gk < K) ? BT[(size_t)gn * K + gk] : (_Float16)0.0f;
      }
    }
    if (k0 + TKK < K) {   // prefetch next K tile (global_prefetch_b8)
      __builtin_prefetch((const void*)&A[(size_t)(bm + (tid & 63)) * K + k0 + TKK], 0, 1);
      __builtin_prefetch((const void*)&BT[(size_t)(bn + (tid & 63)) * K + k0 + TKK], 0, 1);
    }
    asm volatile("s_wait_asynccnt 0x0" ::: "memory");
    __syncthreads();

    // ---- fragments per ISA 7.12.2 (contiguous 16B runs) ----
    // A frag element e -> K = grp*8 + e (e<8) / 16 + grp*8 + (e-8)
    // B frag element e -> K = grp*8 + e (e<8) / 16 + grp*8 + (e-8)
    const v8h a0lo = *(const v8h*)&As[wm + r16][grp * 8];
    const v8h a0hi = *(const v8h*)&As[wm + r16][16 + grp * 8];
    const v8h a1lo = *(const v8h*)&As[wm + 32 + r16][grp * 8];
    const v8h a1hi = *(const v8h*)&As[wm + 32 + r16][16 + grp * 8];
    const v8h blo  = *(const v8h*)&Bs[wn + r16][grp * 8];
    const v8h bhi  = *(const v8h*)&Bs[wn + r16][16 + grp * 8];
    v16h a0 = __builtin_shufflevector(a0lo, a0hi, 0,1,2,3,4,5,6,7,8,9,10,11,12,13,14,15);
    v16h a1 = __builtin_shufflevector(a1lo, a1hi, 0,1,2,3,4,5,6,7,8,9,10,11,12,13,14,15);
    v16h bf = __builtin_shufflevector(blo,  bhi,  0,1,2,3,4,5,6,7,8,9,10,11,12,13,14,15);

    acc0 = __builtin_amdgcn_wmma_f32_16x16x32_f16(false, a0, false, bf,
                                                  (short)0, acc0, false, false);
    acc1 = __builtin_amdgcn_wmma_f32_16x16x32_f16(false, a1, false, bf,
                                                  (short)0, acc1, false, false);
    __syncthreads();
  }

  // C/D layout: VGPR r holds M = r + 8*grp, N = lane%16
#pragma unroll
  for (int r = 0; r < 8; ++r) {
    int n = bn + wn + r16;
    if (n >= N) continue;
    int m0 = bm + wm + grp * 8 + r;
    int m1 = m0 + 32;
    if (m0 < M) C[(size_t)m0 * N + n] = acc0[r];
    if (m1 < M) C[(size_t)m1 * N + n] = acc1[r];
  }
}

// ---- operand conversion: f32 -> f16, A kept MxK row-major, B transposed ----
__global__ void k_cvt_a(_Float16* h, const float* A, int M, int K, int transA) {
  long i = (long)blockIdx.x * blockDim.x + threadIdx.x;
  if (i >= (long)M * K) return;
  int m = (int)(i / K), k = (int)(i - (long)m * K);
  h[i] = (_Float16)(transA ? A[(size_t)k * M + m] : A[i]);
}

__global__ void k_cvt_bT(_Float16* h, const float* B, int N, int K) {
  long i = (long)blockIdx.x * blockDim.x + threadIdx.x;
  if (i >= (long)N * K) return;
  int n = (int)(i / K), k = (int)(i - (long)n * K);
  h[i] = (_Float16)B[(size_t)k * N + n];
}

// ------------------------------ small kernels ------------------------------

__global__ void k_zero(float* p, long n) {
  long i = (long)blockIdx.x * blockDim.x + threadIdx.x;
  long st = (long)gridDim.x * blockDim.x;
  for (; i < n; i += st) p[i] = 0.0f;
}

__global__ void k_copy(float* d, const float* s, long n) {
  long i = (long)blockIdx.x * blockDim.x + threadIdx.x;
  long st = (long)gridDim.x * blockDim.x;
  for (; i < n; i += st) d[i] = s[i];
}

__global__ void k_scatter_edges(float* A, const int* ei, int E, int n) {
  int e = blockIdx.x * blockDim.x + threadIdx.x;
  if (e < E) A[(size_t)ei[e] * n + ei[E + e]] = 1.0f;
}

// column degree of Ah = A + diag(diagfix); dinv = deg>0 ? rsqrt(deg) : 0
__global__ void k_coldeg(const float* A, int n, float fill,
                         float* dinv, float* dfix) {
  int j = blockIdx.x * blockDim.x + threadIdx.x;
  if (j >= n) return;
  float s = 0.f;
  for (int i = 0; i < n; ++i) s += A[(size_t)i * n + j];
  float f = (A[(size_t)j * n + j] == 0.0f) ? fill : 0.0f;
  float deg = s + f;
  dfix[j] = f;
  dinv[j] = (deg > 0.0f) ? rsqrtf(deg) : 0.0f;
}

__global__ void k_scale_rows(float* z, const float* xw, const float* dinv, long nH) {
  long i = (long)blockIdx.x * blockDim.x + threadIdx.x;
  if (i < nH) z[i] = xw[i] * dinv[i / HDIM];
}

// out = dinv[i]*(agg + dfix[i]*z) + b[j] ; optional mish
__global__ void k_gcn_epi(float* out, const float* agg, const float* z,
                          const float* dinv, const float* dfix,
                          const float* b, long nH, int domish) {
  long i = (long)blockIdx.x * blockDim.x + threadIdx.x;
  if (i >= nH) return;
  long r = i / HDIM;
  int  c = (int)(i - r * HDIM);
  float v = dinv[r] * (agg[i] + dfix[r] * z[i]) + b[c];
  out[i] = domish ? mishf(v) : v;
}

__global__ __launch_bounds__(HDIM) void k_ln_mish(const float* x, const float* g,
                                                  const float* b, float* y) {
  __shared__ float red[HDIM];
  int row = blockIdx.x, t = threadIdx.x;
  float v = x[(size_t)row * HDIM + t];
  red[t] = v;
  __syncthreads();
  for (int s = HDIM / 2; s > 0; s >>= 1) {
    if (t < s) red[t] += red[t + s];
    __syncthreads();
  }
  float mean = red[0] / HDIM;
  __syncthreads();
  float d = v - mean;
  red[t] = d * d;
  __syncthreads();
  for (int s = HDIM / 2; s > 0; s >>= 1) {
    if (t < s) red[t] += red[t + s];
    __syncthreads();
  }
  float var = red[0] / HDIM;
  float o = d * rsqrtf(var + 1e-5f) * g[t] + b[t];
  y[(size_t)row * HDIM + t] = mishf(o);
}

// score[i] = tanh( dot(x_i, w) / ||w|| )
__global__ __launch_bounds__(HDIM) void k_score(const float* x, const float* w,
                                                float* s, int n) {
  __shared__ float rd[HDIM], rw[HDIM];
  int row = blockIdx.x, t = threadIdx.x;
  float xv = x[(size_t)row * HDIM + t], wv = w[t];
  rd[t] = xv * wv;
  rw[t] = wv * wv;
  __syncthreads();
  for (int k = HDIM / 2; k > 0; k >>= 1) {
    if (t < k) { rd[t] += rd[t + k]; rw[t] += rw[t + k]; }
    __syncthreads();
  }
  if (t == 0) s[row] = tanhf(rd[0] * rsqrtf(rw[0]));
}

// bitonic top-k: descending score, ascending index on ties (matches top_k)
__global__ __launch_bounds__(1024) void k_sort_topk(const float* score, int n,
                                                    int k, int* perm, float* vals) {
  __shared__ unsigned long long keys[4096];
  int t = threadIdx.x;
  for (int i = t; i < n; i += 1024) {
    unsigned int u = __float_as_uint(score[i]);
    u = (u & 0x80000000u) ? ~u : (u | 0x80000000u);  // order-preserving map
    keys[i] = ((unsigned long long)u << 32) |
              (unsigned long long)(0xFFFFFFFFu - (unsigned)i);
  }
  __syncthreads();
  for (int sz = 2; sz <= n; sz <<= 1) {
    for (int st = sz >> 1; st > 0; st >>= 1) {
      for (int i = t; i < n; i += 1024) {
        int j = i ^ st;
        if (j > i) {
          bool desc = ((i & sz) == 0);
          unsigned long long a = keys[i], b = keys[j];
          bool sw = desc ? (a < b) : (a > b);
          if (sw) { keys[i] = b; keys[j] = a; }
        }
      }
      __syncthreads();
    }
  }
  for (int r = t; r < k; r += 1024) {
    int idx = (int)(0xFFFFFFFFu - (unsigned int)(keys[r] & 0xFFFFFFFFull));
    perm[r] = idx;
    vals[r] = score[idx];
  }
}

__global__ void k_pool_x(float* out, const float* x, const int* perm,
                         const float* vals, long kH) {
  long i = (long)blockIdx.x * blockDim.x + threadIdx.x;
  if (i >= kH) return;
  long r = i / HDIM;
  int  j = (int)(i - r * HDIM);
  out[i] = x[(size_t)perm[r] * HDIM + j] * vals[r];
}

__global__ void k_pool_A(float* Ap, const float* Aaug, const int* perm,
                         int k, int n) {
  long i = (long)blockIdx.x * blockDim.x + threadIdx.x;
  if (i >= (long)k * k) return;
  int r = (int)(i / k), c = (int)(i % k);
  Ap[i] = Aaug[(size_t)perm[r] * n + perm[c]];
}

__global__ void k_make_ai(float* Ai, const float* A, int n) {
  long i = (long)blockIdx.x * blockDim.x + threadIdx.x;
  long st = (long)gridDim.x * blockDim.x;
  long nn = (long)n * n;
  for (; i < nn; i += st) {
    int r = (int)(i / n), c = (int)(i % n);
    Ai[i] = (r == c) ? 1.0f : A[i];
  }
}

__global__ void k_zero_diag(float* M, int n) {
  int i = blockIdx.x * blockDim.x + threadIdx.x;
  if (i < n) M[(size_t)i * n + i] = 0.0f;
}

__global__ void k_scatter_add(float* y, const float* x, const int* perm, long kH) {
  long i = (long)blockIdx.x * blockDim.x + threadIdx.x;
  if (i >= kH) return;
  long r = i / HDIM;
  int  j = (int)(i - r * HDIM);
  y[(size_t)perm[r] * HDIM + j] += x[i];
}

__global__ void k_concat(float* xin, const float* Fc, const float* bf,
                         const float* mesh, int t) {
  int i = blockIdx.x * blockDim.x + threadIdx.x;
  if (i >= NNODE) return;
  float* o = xin + (size_t)i * INDIM;
  o[0] = Fc[i * NFLD + 0];
  o[1] = Fc[i * NFLD + 1];
  o[2] = Fc[i * NFLD + 2];
  o[3] = bf[i * NTIME + t];
#pragma unroll
  for (int m = 0; m < NMESH; ++m) o[4 + m] = mesh[i * NMESH + m];
}

__global__ void k_bias_act(float* y, const float* b, long n, int w, int domish) {
  long i = (long)blockIdx.x * blockDim.x + threadIdx.x;
  if (i >= n) return;
  float v = y[i] + b[i % w];
  y[i] = domish ? mishf(v) : v;
}

__global__ void k_update(float* Fc, const float* Fdot, float* out, int t) {
  int i = blockIdx.x * blockDim.x + threadIdx.x;
  if (i >= NNODE * NFLD) return;
  int node = i / NFLD, f = i % NFLD;
  float v = Fc[i] + TSTEP * Fdot[i];
  out[((size_t)node * (NTIME - 1) + t) * NFLD + f] = v;
  Fc[i] = v;
}

// ------------------------------ host helpers -------------------------------

static inline dim3 gs(long n) { return dim3((unsigned)((n + 255) / 256)); }

struct Scratch {
  float *xw, *z, *agg, *dinv, *dfix;
  _Float16 *h16a, *h16b;
};

static void gemm(hipStream_t s, const float* A, const float* B, float* C,
                 int M, int N, int K, int transA, const Scratch& sc) {
  k_cvt_a <<<gs((long)M * K), 256, 0, s>>>(sc.h16a, A, M, K, transA);
  k_cvt_bT<<<gs((long)N * K), 256, 0, s>>>(sc.h16b, B, N, K);
  dim3 g((N + TN - 1) / TN, (M + TM - 1) / TM);
  k_gemm<<<g, 256, 0, s>>>(sc.h16a, sc.h16b, C, M, N, K);
}

// out = GCN(x, A, W, b); W is kin x H. fill=1 (improved=False) or 2 (True).
static void run_gcn(hipStream_t s, const float* A, int n, const float* x, int kin,
                    const float* W, const float* b, float* out, float fill,
                    int domish, const Scratch& sc) {
  k_coldeg<<<gs(n), 256, 0, s>>>(A, n, fill, sc.dinv, sc.dfix);
  gemm(s, x, W, sc.xw, n, HDIM, kin, 0, sc);
  long nH = (long)n * HDIM;
  k_scale_rows<<<gs(nH), 256, 0, s>>>(sc.z, sc.xw, sc.dinv, nH);
  gemm(s, A, sc.z, sc.agg, n, HDIM, n, 1, sc);  // Ah^T z = A^T z + dfix*z
  k_gcn_epi<<<gs(nH), 256, 0, s>>>(out, sc.agg, sc.z, sc.dinv, sc.dfix, b, nH, domish);
}

// Aout = offdiag( (offdiag(A)+I) @ (offdiag(A)+I) )
static void run_augment(hipStream_t s, const float* A, int n, float* Ai, float* Aout,
                        const Scratch& sc) {
  k_make_ai<<<4096, 256, 0, s>>>(Ai, A, n);
  gemm(s, Ai, Ai, Aout, n, n, n, 0, sc);
  k_zero_diag<<<gs(n), 256, 0, s>>>(Aout, n);
}

// ------------------------------ entry point --------------------------------

extern "C" void kernel_launch(void* const* d_in, const int* in_sizes, int n_in,
                              void* d_out, int out_size, void* d_ws, size_t ws_size,
                              hipStream_t stream) {
  (void)n_in; (void)out_size; (void)ws_size;

  // setup_inputs() insertion order, nested dicts/lists flattened in order:
  const float* F      = (const float*)d_in[0];
  const float* mesh0  = (const float*)d_in[1];
  const float* bfield = (const float*)d_in[2];
  const float* Wpre1  = (const float*)d_in[3];
  const float* bpre1  = (const float*)d_in[4];
  const float* lng    = (const float*)d_in[5];
  const float* lnb    = (const float*)d_in[6];
  const float* Wpre2  = (const float*)d_in[7];
  const float* bpre2  = (const float*)d_in[8];
  const float* Wd[4]  = {(const float*)d_in[9],  (const float*)d_in[10],
                         (const float*)d_in[11], (const float*)d_in[12]};
  const float* bd[4]  = {(const float*)d_in[13], (const float*)d_in[14],
                         (const float*)d_in[15], (const float*)d_in[16]};
  const float* pw[3]  = {(const float*)d_in[17], (const float*)d_in[18],
                         (const float*)d_in[19]};
  const float* Wu[3]  = {(const float*)d_in[20], (const float*)d_in[21],
                         (const float*)d_in[22]};
  const float* bu[3]  = {(const float*)d_in[23], (const float*)d_in[24],
                         (const float*)d_in[25]};
  const float* Wpost1 = (const float*)d_in[26];
  const float* bpost1 = (const float*)d_in[27];
  const float* Wpost2 = (const float*)d_in[28];
  const float* bpost2 = (const float*)d_in[29];
  const float* L1 = (const float*)d_in[30]; const float* bL1 = (const float*)d_in[31];
  const float* L2 = (const float*)d_in[32]; const float* bL2 = (const float*)d_in[33];
  const float* L3 = (const float*)d_in[34]; const float* bL3 = (const float*)d_in[35];
  const int*   ei = (const int*)d_in[36];
  const int E = in_sizes[36] / 2;            // 65536
  float* out = (float*)d_out;

  // ---- workspace bump allocator (~330 MB) ----
  size_t off = 0;
  auto allocb = [&](size_t bytes) {
    void* p = (char*)d_ws + off;
    off = (off + bytes + 255) & ~(size_t)255;
    return p;
  };
  auto alloc = [&](size_t elems) { return allocb(elems * sizeof(float)); };

  float* A0   = (float*)alloc((size_t)NNODE * NNODE);
  float* Ai   = (float*)alloc((size_t)NNODE * NNODE);
  float* Aaug = (float*)alloc((size_t)NNODE * NNODE);
  float* A1p  = (float*)alloc((size_t)2048 * 2048);
  float* A2p  = (float*)alloc((size_t)1024 * 1024);
  float* A3p  = (float*)alloc((size_t)512 * 512);
  float* xs0  = (float*)alloc((size_t)NNODE * HDIM);
  float* xs1  = (float*)alloc((size_t)2048 * HDIM);
  float* xs2  = (float*)alloc((size_t)1024 * HDIM);
  float* xcur = (float*)alloc((size_t)NNODE * HDIM);
  float* xtmp = (float*)alloc((size_t)NNODE * HDIM);
  float* xres = (float*)alloc((size_t)NNODE * HDIM);
  Scratch sc;
  sc.xw   = (float*)alloc((size_t)NNODE * HDIM);
  sc.z    = (float*)alloc((size_t)NNODE * HDIM);
  sc.agg  = (float*)alloc((size_t)NNODE * HDIM);
  sc.dinv = (float*)alloc(NNODE);
  sc.dfix = (float*)alloc(NNODE);
  sc.h16a = (_Float16*)allocb((size_t)NNODE * NNODE * sizeof(_Float16));
  sc.h16b = (_Float16*)allocb((size_t)NNODE * NNODE * sizeof(_Float16));
  float* xin   = (float*)alloc((size_t)NNODE * INDIM);
  float* Fcur  = (float*)alloc((size_t)NNODE * NFLD);
  float* Fdot  = (float*)alloc((size_t)NNODE * NFLD);
  float* score = (float*)alloc(NNODE);
  float* vals  = (float*)alloc(2048);
  int* perm1 = (int*)alloc(2048);
  int* perm2 = (int*)alloc(1024);
  int* perm3 = (int*)alloc(512);

  // ---- build dense adjacency from edge_index (A[s,t]=1) ----
  k_zero<<<4096, 256, 0, stream>>>(A0, (long)NNODE * NNODE);
  k_scatter_edges<<<gs(E), 256, 0, stream>>>(A0, ei, E, NNODE);
  k_copy<<<gs(NNODE * NFLD), 256, 0, stream>>>(Fcur, F, (long)NNODE * NFLD);

  const int pooled[4] = {NNODE, 2048, 1024, 512};
  float* Alev[4] = {A0, A1p, A2p, A3p};
  int* perms[3] = {perm1, perm2, perm3};
  const float* pws[3] = {pw[0], pw[1], pw[2]};
  float* xsv[3] = {xs0, xs1, xs2};

  for (int t = 0; t < NTIME - 1; ++t) {
    // pre_net: GCN -> LayerNorm -> Mish -> GCN -> Mish
    k_concat<<<gs(NNODE), 256, 0, stream>>>(xin, Fcur, bfield, mesh0, t);
    run_gcn(stream, A0, NNODE, xin, INDIM, Wpre1, bpre1, xtmp, 1.0f, 0, sc);
    k_ln_mish<<<NNODE, HDIM, 0, stream>>>(xtmp, lng, lnb, xcur);
    run_gcn(stream, A0, NNODE, xcur, HDIM, Wpre2, bpre2, xtmp, 1.0f, 1, sc);

    // ---- GraphUNet down path ----
    run_gcn(stream, A0, NNODE, xtmp, HDIM, Wd[0], bd[0], xs0, 2.0f, 1, sc);
    const float* xlev = xs0;
    for (int i = 1; i <= DEPTH; ++i) {
      int n = pooled[i - 1], k = pooled[i];
      run_augment(stream, Alev[i - 1], n, Ai, Aaug, sc);
      k_score<<<n, HDIM, 0, stream>>>(xlev, pws[i - 1], score, n);
      k_sort_topk<<<1, 1024, 0, stream>>>(score, n, k, perms[i - 1], vals);
      k_pool_x<<<gs((long)k * HDIM), 256, 0, stream>>>(xcur, xlev, perms[i - 1],
                                                       vals, (long)k * HDIM);
      k_pool_A<<<gs((long)k * k), 256, 0, stream>>>(Alev[i], Aaug, perms[i - 1], k, n);
      float* xdst = (i < DEPTH) ? xsv[i] : xtmp;
      run_gcn(stream, Alev[i], k, xcur, HDIM, Wd[i], bd[i], xdst, 2.0f, 1, sc);
      xlev = xdst;
    }

    // ---- up path (sum_res=True) ----
    for (int i = 0; i < DEPTH; ++i) {
      int j = DEPTH - 1 - i;                 // res level
      int nres = pooled[j], kchild = pooled[j + 1];
      k_copy<<<gs((long)nres * HDIM), 256, 0, stream>>>(xres, xsv[j],
                                                        (long)nres * HDIM);
      k_scatter_add<<<gs((long)kchild * HDIM), 256, 0, stream>>>(
          xres, xtmp, perms[j], (long)kchild * HDIM);
      int domish = (i < DEPTH - 1) ? 1 : 0;
      run_gcn(stream, Alev[j], nres, xres, HDIM, Wu[i], bu[i], xtmp, 2.0f, domish, sc);
    }

    // post_net: GCN Mish GCN Mish Lin Mish Lin Mish Lin
    run_gcn(stream, A0, NNODE, xtmp, HDIM, Wpost1, bpost1, xcur, 1.0f, 1, sc);
    run_gcn(stream, A0, NNODE, xcur, HDIM, Wpost2, bpost2, xtmp, 1.0f, 1, sc);
    gemm(stream, xtmp, L1, xcur, NNODE, HDIM, HDIM, 0, sc);
    k_bias_act<<<gs((long)NNODE * HDIM), 256, 0, stream>>>(xcur, bL1,
                                                           (long)NNODE * HDIM, HDIM, 1);
    gemm(stream, xcur, L2, xtmp, NNODE, HDIM, HDIM, 0, sc);
    k_bias_act<<<gs((long)NNODE * HDIM), 256, 0, stream>>>(xtmp, bL2,
                                                           (long)NNODE * HDIM, HDIM, 1);
    gemm(stream, xtmp, L3, Fdot, NNODE, NFLD, HDIM, 0, sc);
    k_bias_act<<<gs((long)NNODE * NFLD), 256, 0, stream>>>(Fdot, bL3,
                                                           (long)NNODE * NFLD, NFLD, 0);
    k_update<<<gs(NNODE * NFLD), 256, 0, stream>>>(Fcur, Fdot, out, t);
  }
}
